// RelFuseNet_4853313045296
// MI455X (gfx1250) — matrix operand order repaired and checked by reading, not compiled, not run on hardware
//
#include <hip/hip_runtime.h>

// ---------------------------------------------------------------------------
// Types for CDNA5 WMMA (wave32): v_wmma_f32_16x16x32_bf16
// ---------------------------------------------------------------------------
typedef __attribute__((ext_vector_type(16))) __bf16 v16bf;
typedef __attribute__((ext_vector_type(8)))  float  v8f;

union FragU { v16bf v; unsigned int u[8]; };

__device__ inline unsigned short f2bf(float f) {
    unsigned int u = __float_as_uint(f);
    unsigned int r = u + 0x7FFFu + ((u >> 16) & 1u);   // round-to-nearest-even
    return (unsigned short)(r >> 16);
}

// A-fragment (16x32 bf16): lane half holds K = kb..kb+7 and kb+16..kb+23
__device__ inline v16bf load_frag_a(const unsigned short* p) {
    FragU f;
    uint4 t0 = *reinterpret_cast<const uint4*>(p);        // 8 bf16: K = +0..+7
    uint4 t1 = *reinterpret_cast<const uint4*>(p + 16);   // 8 bf16: K = +16..+23
    f.u[0] = t0.x; f.u[1] = t0.y; f.u[2] = t0.z; f.u[3] = t0.w;
    f.u[4] = t1.x; f.u[5] = t1.y; f.u[6] = t1.z; f.u[7] = t1.w;
    return f.v;
}

// B-fragment (32x16 bf16, weights pre-transposed to [Nc,K]):
// lane half holds K = kb..kb+7 and kb+16..kb+23.  Works for global or LDS pointers.
__device__ inline v16bf load_frag_b(const unsigned short* p) {
    FragU f;
    uint4 t0 = *reinterpret_cast<const uint4*>(p);        // K = +0..+7
    uint4 t1 = *reinterpret_cast<const uint4*>(p + 16);   // K = +16..+23
    f.u[0] = t0.x; f.u[1] = t0.y; f.u[2] = t0.z; f.u[3] = t0.w;
    f.u[4] = t1.x; f.u[5] = t1.y; f.u[6] = t1.z; f.u[7] = t1.w;
    return f.v;
}

// ---------------------------------------------------------------------------
// Utility kernels
// ---------------------------------------------------------------------------
__global__ void zero_f32_kernel(float* p, long long n) {
    long long i = blockIdx.x * (long long)blockDim.x + threadIdx.x;
    if (i < n) p[i] = 0.0f;
}

// W [K,Nc] f32 (row-major)  ->  Wt [Nc,K] bf16
__global__ void transpose_cvt_kernel(const float* W, unsigned short* Wt, int K, int Nc) {
    long long i = blockIdx.x * (long long)blockDim.x + threadIdx.x;
    if (i >= (long long)K * Nc) return;
    int k = (int)(i / Nc);
    int n = (int)(i % Nc);
    Wt[(long long)n * K + k] = f2bf(W[i]);
}

__global__ void build_xcat_kernel(const float* hv, const float* ht, const float* htab,
                                  unsigned short* xcat, int N) {
    long long i = blockIdx.x * (long long)blockDim.x + threadIdx.x;
    if (i >= (long long)N * 384) return;
    int n = (int)(i / 384);
    int f = (int)(i % 384);
    float x;
    if (f < 128)      x = hv[(long long)n * 128 + f];
    else if (f < 256) x = ht[(long long)n * 128 + (f - 128)];
    else              x = htab[(long long)n * 128 + (f - 256)];
    xcat[i] = f2bf(x);
}

__global__ void mean_cvt_kernel(const float* agg, const float* cnt, unsigned short* mean,
                                int N, int D) {
    long long i = blockIdx.x * (long long)blockDim.x + threadIdx.x;
    if (i >= (long long)N * D) return;
    int n = (int)(i / D);
    float c = cnt[n];
    if (c < 1.0f) c = 1.0f;
    mean[i] = f2bf(agg[i] / c);
}

// ---------------------------------------------------------------------------
// Edge aggregation: gathers from 3 concatenated f32 feature blocks (D = 384)
// ---------------------------------------------------------------------------
__global__ void aggregate1_kernel(const long long* ei, const float* hv, const float* ht,
                                  const float* htab, float* agg, float* cnt, int E) {
    long long idx = blockIdx.x * (long long)blockDim.x + threadIdx.x;
    long long e = idx / 96;
    int c = (int)(idx % 96);
    if (e >= E) return;
    long long src = ei[e];
    long long dst = ei[(long long)E + e];
    int f = c * 4;
    const float* base;
    if (f < 128)      base = hv   + src * 128 + f;
    else if (f < 256) base = ht   + src * 128 + (f - 128);
    else              base = htab + src * 128 + (f - 256);
    float4 v = *reinterpret_cast<const float4*>(base);
    float* out = agg + dst * 384 + f;
    atomicAdd(out + 0, v.x);
    atomicAdd(out + 1, v.y);
    atomicAdd(out + 2, v.z);
    atomicAdd(out + 3, v.w);
    if (c == 0) atomicAdd(cnt + dst, 1.0f);
}

// Second aggregation over h_g1 f32 [N,256]
__global__ void aggregate2_kernel(const long long* ei, const float* x, float* agg, int E) {
    long long idx = blockIdx.x * (long long)blockDim.x + threadIdx.x;
    long long e = idx / 64;
    int c = (int)(idx % 64);
    if (e >= E) return;
    long long src = ei[e];
    long long dst = ei[(long long)E + e];
    int f = c * 4;
    float4 v = *reinterpret_cast<const float4*>(x + src * 256 + f);
    float* out = agg + dst * 256 + f;
    atomicAdd(out + 0, v.x);
    atomicAdd(out + 1, v.y);
    atomicAdd(out + 2, v.z);
    atomicAdd(out + 3, v.w);
}

// ---------------------------------------------------------------------------
// WMMA GEMM with async-to-LDS weight staging.
//   C[M,Nc] = A1@W1t^T (+ A2@W2t^T) (+ bias) (ReLU optional)
//   A* : bf16 [M x K] row stride lda*;  W*t : bf16 [Nc x K] (transposed weights)
// Block = 8 waves = 8 consecutive 16-row tiles sharing one 64-col strip.
// Weight chunks (64 x 128 bf16 per matrix) staged into LDS via
// global_load_async_to_lds_b128 (ASYNCcnt) and consumed via ds_load_b128.
// ---------------------------------------------------------------------------
#define KC 128
#define LDSTRIDE (KC + 8)   // bf16 elements; 272B row stride -> conflict-free frags

__global__ void gemm_bf16_wmma(const unsigned short* __restrict__ A1, int lda1,
                               const unsigned short* __restrict__ W1t,
                               const unsigned short* __restrict__ A2, int lda2,
                               const unsigned short* __restrict__ W2t,
                               const float* __restrict__ bias,
                               float* __restrict__ Cf, int ldcf,
                               unsigned short* __restrict__ Cb, int ldcb,
                               int M, int Nc, int K, int relu) {
    __shared__ __attribute__((aligned(16))) unsigned short ldsB[2 * 64 * LDSTRIDE];

    int nstrips = Nc >> 6;
    int tiles_m = M >> 4;
    int group = blockIdx.x / nstrips;        // 8-tile m group
    int strip = blockIdx.x % nstrips;
    int wid  = threadIdx.x >> 5;
    int lane = threadIdx.x & 31;
    int mt = group * 8 + wid;
    bool active = (mt < tiles_m);
    int nb = strip << 6;
    int nlane = lane & 15;
    int half = lane >> 4;

    unsigned short* ldsB1 = ldsB;
    unsigned short* ldsB2 = ldsB + 64 * LDSTRIDE;

    v8f acc[4];
    for (int j = 0; j < 4; ++j)
        for (int i = 0; i < 8; ++i) acc[j][i] = 0.0f;

    int rowA = (mt << 4) + nlane;
    const unsigned short* pa1 = A1 + (long long)rowA * lda1 + half * 8;
    const unsigned short* pa2 = A2 ? (A2 + (long long)rowA * lda2 + half * 8) : nullptr;

    for (int kc = 0; kc < K; kc += KC) {
        // ---- stage 64 x KC bf16 weight chunk(s) into LDS (async, 16B/lane) ----
        for (int q = threadIdx.x; q < 64 * (KC / 8); q += blockDim.x) {
            int row = q >> 4;                  // 16 x 16B chunks per 128-elem row
            int cq  = q & 15;
            unsigned long long g1 =
                (unsigned long long)(W1t + (long long)(nb + row) * K + kc + cq * 8);
            unsigned int l1 =
                (unsigned int)(unsigned long long)(ldsB1 + row * LDSTRIDE + cq * 8);
            asm volatile("global_load_async_to_lds_b128 %0, %1, off"
                         :: "v"(l1), "v"(g1) : "memory");
            if (W2t) {
                unsigned long long g2 =
                    (unsigned long long)(W2t + (long long)(nb + row) * K + kc + cq * 8);
                unsigned int l2 =
                    (unsigned int)(unsigned long long)(ldsB2 + row * LDSTRIDE + cq * 8);
                asm volatile("global_load_async_to_lds_b128 %0, %1, off"
                             :: "v"(l2), "v"(g2) : "memory");
            }
        }
        asm volatile("s_wait_asynccnt 0x0" ::: "memory");
        __syncthreads();

        if (active) {
            for (int k0 = 0; k0 < KC; k0 += 32) {
                v16bf a1 = load_frag_a(pa1 + kc + k0);
                __builtin_prefetch(pa1 + kc + k0 + 64, 0, 3);   // near-cache prefetch
#pragma unroll
                for (int j = 0; j < 4; ++j) {
                    const unsigned short* bp =
                        ldsB1 + ((j << 4) + nlane) * LDSTRIDE + k0 + half * 8;
                    v16bf b1 = load_frag_b(bp);
                    acc[j] = __builtin_amdgcn_wmma_f32_16x16x32_bf16(
                        false, a1, false, b1, (short)0, acc[j], false, false);
                }
                if (A2) {
                    v16bf a2 = load_frag_a(pa2 + kc + k0);
#pragma unroll
                    for (int j = 0; j < 4; ++j) {
                        const unsigned short* bp =
                            ldsB2 + ((j << 4) + nlane) * LDSTRIDE + k0 + half * 8;
                        v16bf b2 = load_frag_b(bp);
                        acc[j] = __builtin_amdgcn_wmma_f32_16x16x32_bf16(
                            false, a2, false, b2, (short)0, acc[j], false, false);
                    }
                }
            }
        }
        __syncthreads();   // protect LDS before next chunk overwrite
    }

    if (!active) return;

    // Epilogue: C layout — lane gives N=lane&15; VGPR i gives M = i + half*8
    for (int j = 0; j < 4; ++j) {
        int col = nb + (j << 4) + nlane;
        float bv = bias ? bias[col] : 0.0f;
#pragma unroll
        for (int i = 0; i < 8; ++i) {
            int row = (mt << 4) + i + half * 8;
            float x = acc[j][i] + bv;
            if (relu) x = fmaxf(x, 0.0f);
            if (Cf) Cf[(long long)row * ldcf + col] = x;
            if (Cb) Cb[(long long)row * ldcb + col] = f2bf(x);
        }
    }
}

// ---------------------------------------------------------------------------
// Row L2-normalize (wave per row); also writes bf16 into combined[:,384:512]
// ---------------------------------------------------------------------------
__global__ void normalize_kernel(float* g2, unsigned short* comb, int N) {
    int wid = (int)((blockIdx.x * (long long)blockDim.x + threadIdx.x) >> 5);
    int lane = threadIdx.x & 31;
    if (wid >= N) return;
    float* r = g2 + (long long)wid * 128;
    float v[4];
    float ss = 0.0f;
#pragma unroll
    for (int i = 0; i < 4; ++i) { v[i] = r[lane + 32 * i]; ss += v[i] * v[i]; }
    for (int m = 16; m >= 1; m >>= 1) ss += __shfl_xor(ss, m, 32);
    float inv = 1.0f / fmaxf(sqrtf(ss), 1e-12f);
    unsigned short* cb = comb + (long long)wid * 512 + 384;
#pragma unroll
    for (int i = 0; i < 4; ++i) {
        float x = v[i] * inv;
        r[lane + 32 * i] = x;
        cb[lane + 32 * i] = f2bf(x);
    }
}

// ---------------------------------------------------------------------------
// Final tiny GEMM: logits[N,14] = hid[N,256] @ Wc2[256,14] + bc2 (wave per row)
// ---------------------------------------------------------------------------
__global__ void logits_kernel(const float* __restrict__ hid, const float* __restrict__ Wc2,
                              const float* __restrict__ bc2, float* __restrict__ out, int N) {
    int wid = (int)((blockIdx.x * (long long)blockDim.x + threadIdx.x) >> 5);
    int lane = threadIdx.x & 31;
    if (wid >= N) return;
    const float* h = hid + (long long)wid * 256;
    float s[14];
#pragma unroll
    for (int c = 0; c < 14; ++c) s[c] = 0.0f;
    for (int k = lane; k < 256; k += 32) {
        float hv = h[k];
        const float* w = Wc2 + (long long)k * 14;
#pragma unroll
        for (int c = 0; c < 14; ++c) s[c] += hv * w[c];
    }
#pragma unroll
    for (int c = 0; c < 14; ++c)
        for (int m = 16; m >= 1; m >>= 1) s[c] += __shfl_xor(s[c], m, 32);
    if (lane == 0) {
        float* o = out + (long long)wid * 14;
#pragma unroll
        for (int c = 0; c < 14; ++c) o[c] = s[c] + bc2[c];
    }
}

// ---------------------------------------------------------------------------
// Host launch
// ---------------------------------------------------------------------------
extern "C" void kernel_launch(void* const* d_in, const int* in_sizes, int n_in,
                              void* d_out, int out_size, void* d_ws, size_t ws_size,
                              hipStream_t stream) {
    (void)n_in; (void)out_size; (void)ws_size;

    const float*      h_v   = (const float*)d_in[0];
    const float*      h_t   = (const float*)d_in[1];
    const float*      h_tab = (const float*)d_in[2];
    const long long*  ei    = (const long long*)d_in[3];
    const float* W1l  = (const float*)d_in[4];
    const float* W1r  = (const float*)d_in[5];
    const float* b1g  = (const float*)d_in[6];
    const float* W2l  = (const float*)d_in[7];
    const float* W2r  = (const float*)d_in[8];
    const float* b2g  = (const float*)d_in[9];
    const float* Wsh  = (const float*)d_in[10];
    const float* bsh  = (const float*)d_in[11];
    const float* Wst  = (const float*)d_in[12];
    const float* bst  = (const float*)d_in[13];
    const float* Wstab= (const float*)d_in[14];
    const float* bstab= (const float*)d_in[15];
    const float* Wc1  = (const float*)d_in[16];
    const float* bc1  = (const float*)d_in[17];
    const float* Wc2  = (const float*)d_in[18];
    const float* bc2  = (const float*)d_in[19];

    const int N = in_sizes[0] / 128;
    const int E = in_sizes[3] / 2;

    // ---- workspace layout (aliased where lifetimes permit) ----
    char* ws = (char*)d_ws;
    size_t off = 0;
    auto take = [&](size_t bytes) -> char* {
        char* p = ws + off;
        off += (bytes + 255) & ~(size_t)255;
        return p;
    };
    float*          agg   = (float*)take((size_t)N * 384 * 4);   // also agg2 [N,256]
    float*          cnt   = (float*)take((size_t)N * 4);
    unsigned short* xcat  = (unsigned short*)take((size_t)N * 384 * 2);
    unsigned short* mean1 = (unsigned short*)take((size_t)N * 384 * 2); // also mean2 [N,256]
    float*          g1f   = (float*)take((size_t)N * 256 * 4);   // also hid [N,256]
    unsigned short* g1b   = (unsigned short*)take((size_t)N * 256 * 2);
    float*          g2f   = (float*)take((size_t)N * 128 * 4);
    unsigned short* comb  = (unsigned short*)take((size_t)N * 512 * 2);
    unsigned short* w1lt  = (unsigned short*)take((size_t)384 * 256 * 2);
    unsigned short* w1rt  = (unsigned short*)take((size_t)384 * 256 * 2);
    unsigned short* w2lt  = (unsigned short*)take((size_t)256 * 128 * 2);
    unsigned short* w2rt  = (unsigned short*)take((size_t)256 * 128 * 2);
    unsigned short* wsht  = (unsigned short*)take((size_t)128 * 128 * 2);
    unsigned short* wstt  = (unsigned short*)take((size_t)128 * 128 * 2);
    unsigned short* wstabt= (unsigned short*)take((size_t)128 * 128 * 2);
    unsigned short* wc1t  = (unsigned short*)take((size_t)512 * 256 * 2);
    float*          agg2  = agg;
    unsigned short* mean2 = mean1;
    float*          hid   = g1f;

    float* out_logits = (float*)d_out;
    float* out_zsh    = out_logits + (size_t)N * 14;
    float* out_zt     = out_zsh + (size_t)N * 128;
    float* out_ztab   = out_zt  + (size_t)N * 128;

    auto blocks = [](long long n) { return (unsigned)((n + 255) / 256); };
    auto launch_gemm = [&](const unsigned short* A1, int lda1, const unsigned short* W1t,
                           const unsigned short* A2, int lda2, const unsigned short* W2t,
                           const float* bias, float* Cf, int ldcf,
                           unsigned short* Cb, int ldcb, int M, int Nc, int K, int relu) {
        int tiles_m = M / 16;
        int mgroups = (tiles_m + 7) / 8;
        int nstrips = Nc / 64;
        gemm_bf16_wmma<<<(unsigned)(mgroups * nstrips), 256, 0, stream>>>(
            A1, lda1, W1t, A2, lda2, W2t, bias, Cf, ldcf, Cb, ldcb, M, Nc, K, relu);
    };

    // 1) zero accumulators
    zero_f32_kernel<<<blocks((long long)N * 384), 256, 0, stream>>>(agg, (long long)N * 384);
    zero_f32_kernel<<<blocks(N), 256, 0, stream>>>(cnt, N);

    // 2) weights -> bf16 transposed [Nc,K]
    transpose_cvt_kernel<<<blocks(384 * 256), 256, 0, stream>>>(W1l, w1lt, 384, 256);
    transpose_cvt_kernel<<<blocks(384 * 256), 256, 0, stream>>>(W1r, w1rt, 384, 256);
    transpose_cvt_kernel<<<blocks(256 * 128), 256, 0, stream>>>(W2l, w2lt, 256, 128);
    transpose_cvt_kernel<<<blocks(256 * 128), 256, 0, stream>>>(W2r, w2rt, 256, 128);
    transpose_cvt_kernel<<<blocks(128 * 128), 256, 0, stream>>>(Wsh, wsht, 128, 128);
    transpose_cvt_kernel<<<blocks(128 * 128), 256, 0, stream>>>(Wst, wstt, 128, 128);
    transpose_cvt_kernel<<<blocks(128 * 128), 256, 0, stream>>>(Wstab, wstabt, 128, 128);
    transpose_cvt_kernel<<<blocks(512 * 256), 256, 0, stream>>>(Wc1, wc1t, 512, 256);

    // 3) bf16 concat of node features
    build_xcat_kernel<<<blocks((long long)N * 384), 256, 0, stream>>>(h_v, h_t, h_tab, xcat, N);

    // 4) SAGE1 aggregation (f32 atomics) + degree
    aggregate1_kernel<<<blocks((long long)E * 96), 256, 0, stream>>>(ei, h_v, h_t, h_tab, agg, cnt, E);

    // 5) mean -> bf16
    mean_cvt_kernel<<<blocks((long long)N * 384), 256, 0, stream>>>(agg, cnt, mean1, N, 384);

    // 6) SAGE1 GEMM: g1 = relu(mean@W1l + xcat@W1r + b1g), f32 + bf16 outputs
    launch_gemm(mean1, 384, w1lt, xcat, 384, w1rt, b1g, g1f, 256, g1b, 256, N, 256, 384, 1);

    // 7) SAGE2 aggregation
    zero_f32_kernel<<<blocks((long long)N * 256), 256, 0, stream>>>(agg2, (long long)N * 256);
    aggregate2_kernel<<<blocks((long long)E * 64), 256, 0, stream>>>(ei, g1f, agg2, E);
    mean_cvt_kernel<<<blocks((long long)N * 256), 256, 0, stream>>>(agg2, cnt, mean2, N, 256);

    // 8) SAGE2 GEMM: g2 = mean2@W2l + g1@W2r + b2g
    launch_gemm(mean2, 256, w2lt, g1b, 256, w2rt, b2g, g2f, 128, nullptr, 0, N, 128, 256, 0);

    // 9) L2 normalize rows; bf16 copy into combined[:,384:512]
    normalize_kernel<<<blocks((long long)N * 32), 256, 0, stream>>>(g2f, comb, N);

    // 10) projections: f32 into d_out, bf16 into combined columns
    launch_gemm(comb + 384, 512, wsht, nullptr, 0, nullptr, bsh, out_zsh, 128, comb + 0,   512, N, 128, 128, 0);
    launch_gemm(xcat + 128, 384, wstt, nullptr, 0, nullptr, bst, out_zt,  128, comb + 128, 512, N, 128, 128, 0);
    launch_gemm(xcat + 256, 384, wstabt, nullptr, 0, nullptr, bstab, out_ztab, 128, comb + 256, 512, N, 128, 128, 0);

    // 11) classifier hidden: hid = relu(combined@Wc1 + bc1)
    launch_gemm(comb, 512, wc1t, nullptr, 0, nullptr, bc1, hid, 256, nullptr, 0, N, 256, 512, 1);

    // 12) logits
    logits_kernel<<<blocks((long long)N * 32), 256, 0, stream>>>(hid, Wc2, bc2, out_logits, N);
}
